// Encoder_4724464025749
// MI455X (gfx1250) — matrix operand order repaired and checked by reading, not compiled, not run on hardware
//
#include <hip/hip_runtime.h>
#include <hip/hip_bf16.h>

// ---------------------------------------------------------------------------
// Tree-GRU encoder for MI455X (gfx1250, wave32, WMMA f32<-f16 16x16x32).
// M=32 rows per workgroup: each B fragment is reused by two WMMAs, halving
// L2 B-operand traffic vs an M=16 tile. Hidden states carried in f16.
// ---------------------------------------------------------------------------

typedef __attribute__((ext_vector_type(16))) _Float16 v16h;
typedef __attribute__((ext_vector_type(8)))  float    v8f;

#define HDIM  256
#define IDIM  32
#define ODIM  128
#define BATCH 4096
#define TDEPTH 7

union AF { v16h v; _Float16 h[16]; };

// A-fragment (16x32 f16) gather from row-major LDS, per ISA layout:
// lane L: row = L&15, hi = L>>4; halves 0..7 -> K = k0+hi*8+j,
// halves 8..15 -> K = k0+hi*8+16+j.  (two contiguous 16B runs -> ds_load_b128)
__device__ inline v16h ldsA(const _Float16* base, int stride, int lane, int k0) {
    const int row = lane & 15, hi = (lane >> 4) & 1;
    const _Float16* p = base + row * stride + k0 + hi * 8;
    AF a;
#pragma unroll
    for (int j = 0; j < 8; ++j) { a.h[j] = p[j]; a.h[j + 8] = p[j + 16]; }
    return a.v;
}

// B-fragment: pre-packed lane-major, one contiguous 32B load per lane.
__device__ inline v16h packB(const _Float16* pack, int lane, int ct, int ks, int KS) {
    const _Float16* p = pack + (((size_t)ct * KS + ks) * 32 + (size_t)lane) * 16;
    AF b;
#pragma unroll
    for (int j = 0; j < 16; ++j) b.h[j] = p[j];
    return b.v;
}

__device__ inline v8f wmma16(v16h a, v16h b, v8f c) {
    return __builtin_amdgcn_wmma_f32_16x16x32_f16(false, a, false, b, (short)0, c,
                                                  false, false);
}

// 32-row x 64-col GEMM slice: two 16-row M tiles x 4 N-tiles per wave.
// Each B fragment feeds two WMMAs (M reuse).
template <int KS>
__device__ inline void gemm2x4(const _Float16* aLds, int stride,
                               const _Float16* pack, int lane, int ct0,
                               v8f acc[2][4]) {
#pragma unroll
    for (int ks = 0; ks < KS; ++ks) {
        if (ks + 1 < KS)  // prefetch next K-step's B block (global_prefetch_b8)
            __builtin_prefetch(pack + (((size_t)ct0 * KS + ks + 1) * 32 + lane) * 16,
                               0, 1);
        v16h a0 = ldsA(aLds,               stride, lane, ks * 32);
        v16h a1 = ldsA(aLds + 16 * stride, stride, lane, ks * 32);
#pragma unroll
        for (int t = 0; t < 4; ++t) {
            v16h b = packB(pack, lane, ct0 + t, ks, KS);
            acc[0][t] = wmma16(a0, b, acc[0][t]);
            acc[1][t] = wmma16(a1, b, acc[1][t]);
        }
    }
}

__device__ inline float sigm(float x)     { return 1.f / (1.f + __expf(-x)); }
__device__ inline float tanhfast(float x) { return 1.f - 2.f / (__expf(2.f * x) + 1.f); }

// ---------------------------------------------------------------------------
// Weight packing: W (N x K, row-major f32) -> f16 fragment blocks.
// pack[(((ct*KS+ks)*32+lane)*16)+j] = W[ct*16 + (lane&15)][ks*32 + (lane>>4)*8 + (j&7) + (j>>3)*16]
// ---------------------------------------------------------------------------
__global__ void pack_kernel(const float* __restrict__ W, _Float16* __restrict__ out,
                            int N, int K) {
    int t = blockIdx.x * blockDim.x + threadIdx.x;
    int total = N * K;
    if (t >= total) return;
    int j = t & 15;
    int lane = (t >> 4) & 31;
    int blk = t >> 9;
    int KS = K / 32;
    int ks = blk % KS;
    int ct = blk / KS;
    int n = ct * 16 + (lane & 15);
    int k = ks * 32 + ((lane >> 4) & 1) * 8 + (j & 7) + (j >> 3) * 16;
    out[t] = (_Float16)W[(size_t)n * K + k];
}

// ---------------------------------------------------------------------------
// Fused per-level kernel: one workgroup = 32 node-batch rows, 4 waves each
// owning a 64-wide slice of H. Children / child_sum / s live in LDS (f16).
// ---------------------------------------------------------------------------
struct LevelArgs {
    const float* targets;
    const float* mask;
    const _Float16 *pWir, *pWiz, *pWin, *pWhr, *pWhz, *pWhn, *pM0, *pM1, *pWa;
    const float *bIr, *bHr, *bIz, *bHz, *bIn, *bHn, *bMs, *wW, *wB, *bWa;
    const _Float16* hPrev;
    _Float16* hCur;
    int start;
    int isLeaf;
};

__global__ __launch_bounds__(128) void level_kernel(LevelArgs A) {
    __shared__ _Float16 sx[32][IDIM];    // x tile (f16)
    __shared__ _Float16 s0[32][HDIM];    // child 0 hidden
    __shared__ _Float16 s1[32][HDIM];    // child 1 hidden
    __shared__ _Float16 sg[32][HDIM];    // child_sum (tanh(Wa g))
    __shared__ _Float16 ssum[32][HDIM];  // s = r0*c0 + r1*c1
    __shared__ float    sc[2][32];       // attention score partials

    const int tid  = threadIdx.x;
    const int lane = tid & 31;
    const int wv   = tid >> 5;
    const int hi   = (lane >> 4) & 1;
    const int colBase = lane & 15;
    const int ct0  = wv * 4;

    const int tilesPerNode = BATCH / 32;
    const int node  = blockIdx.x / tilesPerNode;
    const int b0    = (blockIdx.x % tilesPerNode) * 32;
    const int gnode = A.start + node;

    // ---- load x (f32 -> f16): 32x32, contiguous ----
    const float* xp = A.targets + ((size_t)gnode * BATCH + b0) * IDIM;
    for (int e = tid; e < 32 * IDIM; e += 128)
        sx[e >> 5][e & 31] = (_Float16)xp[e];

    if (A.isLeaf) {
        uint4* z0 = (uint4*)&s0[0][0];
        uint4* z1 = (uint4*)&s1[0][0];
        uint4* z2 = (uint4*)&sg[0][0];
        uint4* z3 = (uint4*)&ssum[0][0];
        const uint4 zz = {0u, 0u, 0u, 0u};
        for (int e = tid; e < (32 * HDIM * 2) / 16; e += 128) {
            z0[e] = zz; z1[e] = zz; z2[e] = zz; z3[e] = zz;
        }
    } else {
        // children of node are contiguous 16KB spans -> b128 copies
        const uint4* c0p =
            (const uint4*)(A.hPrev + ((size_t)(2 * node)     * BATCH + b0) * HDIM);
        const uint4* c1p =
            (const uint4*)(A.hPrev + ((size_t)(2 * node + 1) * BATCH + b0) * HDIM);
        uint4* d0 = (uint4*)&s0[0][0];
        uint4* d1 = (uint4*)&s1[0][0];
        for (int e = tid; e < (32 * HDIM * 2) / 16; e += 128) {
            d0[e] = c0p[e];
            d1[e] = c1p[e];
        }
        if (tid < 64) sc[tid >> 5][tid & 31] = 0.f;
    }
    __syncthreads();

    if (!A.isLeaf) {
        // ---- ms_k = tanh(c_k @ Wms_k^T + b); scores = ms . w_w (ds_add_f32) ----
        v8f m0[2][4], m1[2][4];
#pragma unroll
        for (int mt = 0; mt < 2; ++mt)
#pragma unroll
            for (int t = 0; t < 4; ++t) {
                int col = (ct0 + t) * 16 + colBase;
                float bb0 = A.bMs[col], bb1 = A.bMs[HDIM + col];
#pragma unroll
                for (int v = 0; v < 8; ++v) { m0[mt][t][v] = bb0; m1[mt][t][v] = bb1; }
            }
        gemm2x4<8>(&s0[0][0], HDIM, A.pM0, lane, ct0, m0);
        gemm2x4<8>(&s1[0][0], HDIM, A.pM1, lane, ct0, m1);
#pragma unroll
        for (int mt = 0; mt < 2; ++mt)
#pragma unroll
            for (int v = 0; v < 8; ++v) {
                float p0 = 0.f, p1 = 0.f;
#pragma unroll
                for (int t = 0; t < 4; ++t) {
                    float ww = A.wW[(ct0 + t) * 16 + colBase];
                    p0 += tanhfast(m0[mt][t][v]) * ww;
                    p1 += tanhfast(m1[mt][t][v]) * ww;
                }
                int row = mt * 16 + hi * 8 + v;
                atomicAdd(&sc[0][row], p0);
                atomicAdd(&sc[1][row], p1);
            }
        __syncthreads();

        // ---- a_k = score_k / sum; g = a0*c0+a1*c1; child_sum = tanh(g Wa^T + b) ----
        float wb = A.wB[0];
        float aw0[2], aw1[2];
#pragma unroll
        for (int mt = 0; mt < 2; ++mt) {
            float sA = sc[0][mt * 16 + colBase] + wb;  // A-frag row == lane&15
            float sB = sc[1][mt * 16 + colBase] + wb;
            float inv = 1.f / (sA + sB);
            aw0[mt] = sA * inv;
            aw1[mt] = sB * inv;
        }
        v8f cs[2][4];
#pragma unroll
        for (int mt = 0; mt < 2; ++mt)
#pragma unroll
            for (int t = 0; t < 4; ++t) {
                float bb = A.bWa[(ct0 + t) * 16 + colBase];
#pragma unroll
                for (int v = 0; v < 8; ++v) cs[mt][t][v] = bb;
            }
#pragma unroll
        for (int ks = 0; ks < 8; ++ks) {
            AF af[2];
#pragma unroll
            for (int mt = 0; mt < 2; ++mt) {
                const _Float16* p0 = &s0[mt * 16 + colBase][ks * 32 + hi * 8];
                const _Float16* p1 = &s1[mt * 16 + colBase][ks * 32 + hi * 8];
#pragma unroll
                for (int j = 0; j < 8; ++j) {
                    af[mt].h[j] =
                        (_Float16)(aw0[mt] * (float)p0[j] + aw1[mt] * (float)p1[j]);
                    af[mt].h[j + 8] =
                        (_Float16)(aw0[mt] * (float)p0[j + 16] +
                                   aw1[mt] * (float)p1[j + 16]);
                }
            }
#pragma unroll
            for (int t = 0; t < 4; ++t) {
                v16h b = packB(A.pWa, lane, ct0 + t, ks, 8);
                cs[0][t] = wmma16(af[0].v, b, cs[0][t]);
                cs[1][t] = wmma16(af[1].v, b, cs[1][t]);
            }
        }
#pragma unroll
        for (int mt = 0; mt < 2; ++mt)
#pragma unroll
            for (int t = 0; t < 4; ++t) {
                int col = (ct0 + t) * 16 + colBase;
#pragma unroll
                for (int v = 0; v < 8; ++v)
                    sg[mt * 16 + hi * 8 + v][col] = (_Float16)tanhfast(cs[mt][t][v]);
            }
        __syncthreads();

        // ---- r_k = sigmoid(x Wir^T + c_k Whr^T + b); s = r0*c0 + r1*c1 ----
        v8f r0[2][4], r1[2][4];
#pragma unroll
        for (int mt = 0; mt < 2; ++mt)
#pragma unroll
            for (int t = 0; t < 4; ++t) {
                int col = (ct0 + t) * 16 + colBase;
                float bb = A.bIr[col] + A.bHr[col];
#pragma unroll
                for (int v = 0; v < 8; ++v) { r0[mt][t][v] = bb; r1[mt][t][v] = bb; }
            }
        gemm2x4<1>(&sx[0][0], IDIM, A.pWir, lane, ct0, r0);
        gemm2x4<8>(&s0[0][0], HDIM, A.pWhr, lane, ct0, r0);
        gemm2x4<1>(&sx[0][0], IDIM, A.pWir, lane, ct0, r1);
        gemm2x4<8>(&s1[0][0], HDIM, A.pWhr, lane, ct0, r1);
#pragma unroll
        for (int mt = 0; mt < 2; ++mt)
#pragma unroll
            for (int t = 0; t < 4; ++t) {
                int col = (ct0 + t) * 16 + colBase;
#pragma unroll
                for (int v = 0; v < 8; ++v) {
                    int row = mt * 16 + hi * 8 + v;
                    float rv0 = sigm(r0[mt][t][v]), rv1 = sigm(r1[mt][t][v]);
                    ssum[row][col] = (_Float16)(rv0 * (float)s0[row][col] +
                                                rv1 * (float)s1[row][col]);
                }
            }
        __syncthreads();
    }

    // ---- z = sigmoid(x Wiz + cs Whz + b); n = tanh(x Win + s Whn + b);
    //      h = (1-z)*n + z*child_sum; h *= mask ----
    v8f z[2][4], nn[2][4];
#pragma unroll
    for (int mt = 0; mt < 2; ++mt)
#pragma unroll
        for (int t = 0; t < 4; ++t) {
            int col = (ct0 + t) * 16 + colBase;
            float bz = A.bIz[col] + A.bHz[col];
            float bn = A.bIn[col] + A.bHn[col];
#pragma unroll
            for (int v = 0; v < 8; ++v) { z[mt][t][v] = bz; nn[mt][t][v] = bn; }
        }
    gemm2x4<1>(&sx[0][0],   IDIM, A.pWiz, lane, ct0, z);
    gemm2x4<8>(&sg[0][0],   HDIM, A.pWhz, lane, ct0, z);
    gemm2x4<1>(&sx[0][0],   IDIM, A.pWin, lane, ct0, nn);
    gemm2x4<8>(&ssum[0][0], HDIM, A.pWhn, lane, ct0, nn);

#pragma unroll
    for (int mt = 0; mt < 2; ++mt)
#pragma unroll
        for (int t = 0; t < 4; ++t) {
            int col = (ct0 + t) * 16 + colBase;
#pragma unroll
            for (int v = 0; v < 8; ++v) {
                int row = mt * 16 + hi * 8 + v;
                float zz  = sigm(z[mt][t][v]);
                float nv  = tanhfast(nn[mt][t][v]);
                float csv = (float)sg[row][col];
                float h   = (1.f - zz) * nv + zz * csv;
                float m   = A.mask[(size_t)gnode * BATCH + b0 + row];
                A.hCur[((size_t)node * BATCH + b0 + row) * HDIM + col] =
                    (_Float16)(h * m);
            }
        }
}

// ---------------------------------------------------------------------------
// Head: mu = root @ mu_w^T + b ; lv = root @ lv_w^T + b  (f32 out)
// ---------------------------------------------------------------------------
struct HeadArgs {
    const _Float16* root;
    const _Float16 *pMu, *pLv;
    const float *bMu, *bLv;
    float* out;
};

__global__ __launch_bounds__(128) void head_kernel(HeadArgs A) {
    __shared__ _Float16 sr[16][HDIM];
    const int tid = threadIdx.x, lane = tid & 31, wv = tid >> 5;
    const int hi = (lane >> 4) & 1, colBase = lane & 15;
    const int b0 = blockIdx.x * 16;

    for (int e = tid; e < 16 * HDIM; e += 128) {
        int r = e / HDIM, c = e % HDIM;
        sr[r][c] = A.root[(size_t)(b0 + r) * HDIM + c];
    }
    __syncthreads();

#pragma unroll
    for (int q = 0; q < 4; ++q) {
        int cc = wv * 4 + q;                 // 0..7 -> mu tiles, 8..15 -> lv tiles
        const _Float16* pk = (cc < 8) ? A.pMu : A.pLv;
        const float* bias  = (cc < 8) ? A.bMu : A.bLv;
        int ct = (cc < 8) ? cc : cc - 8;
        v8f acc;
        float bb = bias[ct * 16 + colBase];
#pragma unroll
        for (int v = 0; v < 8; ++v) acc[v] = bb;
#pragma unroll
        for (int ks = 0; ks < 8; ++ks) {
            v16h a = ldsA(&sr[0][0], HDIM, lane, ks * 32);
            acc = wmma16(a, packB(pk, lane, ct, ks, 8), acc);
        }
        size_t outBase = (cc < 8) ? 0 : (size_t)BATCH * ODIM;
#pragma unroll
        for (int v = 0; v < 8; ++v) {
            int row = hi * 8 + v;
            A.out[outBase + (size_t)(b0 + row) * ODIM + ct * 16 + colBase] = acc[v];
        }
    }
}

// ---------------------------------------------------------------------------
extern "C" void kernel_launch(void* const* d_in, const int* in_sizes, int n_in,
                              void* d_out, int out_size, void* d_ws, size_t ws_size,
                              hipStream_t stream) {
    const float* targets = (const float*)d_in[0];
    const float* mask    = (const float*)d_in[1];
    const float* wir_w = (const float*)d_in[2];  const float* wir_b = (const float*)d_in[3];
    const float* whr_w = (const float*)d_in[4];  const float* whr_b = (const float*)d_in[5];
    const float* wiz_w = (const float*)d_in[6];  const float* wiz_b = (const float*)d_in[7];
    const float* whz_w = (const float*)d_in[8];  const float* whz_b = (const float*)d_in[9];
    const float* win_w = (const float*)d_in[10]; const float* win_b = (const float*)d_in[11];
    const float* whn_w = (const float*)d_in[12]; const float* whn_b = (const float*)d_in[13];
    const float* wms_w = (const float*)d_in[14]; const float* wms_b = (const float*)d_in[15];
    const float* w_w   = (const float*)d_in[16]; const float* w_b   = (const float*)d_in[17];
    const float* wa_w  = (const float*)d_in[18]; const float* wa_b  = (const float*)d_in[19];
    const float* mu_w  = (const float*)d_in[20]; const float* mu_b  = (const float*)d_in[21];
    const float* lv_w  = (const float*)d_in[22]; const float* lv_b  = (const float*)d_in[23];

    _Float16* ws = (_Float16*)d_ws;

    // packed-weight layout (offsets in halves)
    size_t off = 0;
    auto take = [&](size_t n) { size_t o = off; off += n; return o; };
    size_t oWir = take(HDIM * IDIM), oWiz = take(HDIM * IDIM), oWin = take(HDIM * IDIM);
    size_t oWhr = take(HDIM * HDIM), oWhz = take(HDIM * HDIM), oWhn = take(HDIM * HDIM);
    size_t oM0  = take(HDIM * HDIM), oM1  = take(HDIM * HDIM), oWa  = take(HDIM * HDIM);
    size_t oMu  = take(ODIM * HDIM), oLv  = take(ODIM * HDIM);
    // ping-pong hidden buffers (f16), after 1 MiB of packed weights
    size_t hid0 = 524288;                                  // halves
    size_t hid1 = hid0 + (size_t)64 * BATCH * HDIM;        // halves

    auto pack = [&](const float* W, size_t o, int N, int K) {
        int total = N * K;
        pack_kernel<<<(total + 255) / 256, 256, 0, stream>>>(W, ws + o, N, K);
    };
    pack(wir_w, oWir, HDIM, IDIM);
    pack(wiz_w, oWiz, HDIM, IDIM);
    pack(win_w, oWin, HDIM, IDIM);
    pack(whr_w, oWhr, HDIM, HDIM);
    pack(whz_w, oWhz, HDIM, HDIM);
    pack(whn_w, oWhn, HDIM, HDIM);
    pack(wms_w,             oM0, HDIM, HDIM);
    pack(wms_w + HDIM*HDIM, oM1, HDIM, HDIM);
    pack(wa_w, oWa, HDIM, HDIM);
    pack(mu_w, oMu, ODIM, HDIM);
    pack(lv_w, oLv, ODIM, HDIM);

    for (int d = TDEPTH - 1; d >= 0; --d) {
        int n = 1 << d;
        int start = n - 1;
        _Float16* cur        = ws + (((d & 1) == 0) ? hid0 : hid1);
        const _Float16* prev = ws + ((((d + 1) & 1) == 0) ? hid0 : hid1);
        LevelArgs A = {targets, mask,
                       ws + oWir, ws + oWiz, ws + oWin,
                       ws + oWhr, ws + oWhz, ws + oWhn,
                       ws + oM0,  ws + oM1,  ws + oWa,
                       wir_b, whr_b, wiz_b, whz_b, win_b, whn_b,
                       wms_b, w_w, w_b, wa_b,
                       prev, cur, start, (d == TDEPTH - 1) ? 1 : 0};
        level_kernel<<<n * (BATCH / 32), 128, 0, stream>>>(A);
    }

    HeadArgs HA = {ws + hid0, ws + oMu, ws + oLv, mu_b, lv_b, (float*)d_out};
    head_kernel<<<BATCH / 16, 128, 0, stream>>>(HA);
}
